// MHA_47347719471536
// MI455X (gfx1250) — matrix operand order repaired and checked by reading, not compiled, not run on hardware
//
#include <hip/hip_runtime.h>
#include <hip/hip_bf16.h>
#include <stdint.h>

// ---------------------------------------------------------------------------
// CDNA5 (gfx1250) fused MHA: QKV GEMM -> flash attention -> output GEMM.
//  - all matmuls: v_wmma_f32_16x16x32_bf16 (wave32 WMMA)
//  - all fragments fed by contiguous 16B LDS reads (ds_load_b128); B operands
//    stored transposed [n][k] so the per-lane K-run is contiguous
//  - K-chunk / bf16 A-tile staging uses GLOBAL_LOAD_ASYNC_TO_LDS_B128
//    (ASYNCcnt) via inline asm; completion via s_wait_asynccnt
//  - LDS leading dims padded (40 / 72) for bank spread, 16B aligned
// ---------------------------------------------------------------------------

typedef __attribute__((ext_vector_type(16))) __bf16 v16bf;
typedef __attribute__((ext_vector_type(8)))  float  v8f;

#define N_HEADS  16
#define HEAD_DIM 64
#define D_MODEL  1024
#define SEQ      2048
#define BATCH    4

#define LDA  40   // padded ld (elems) for 32-wide row-major tiles (80B rows)
#define LDK  72   // padded ld (elems) for 64-wide K tile (144B rows)

__device__ __forceinline__ v8f wmma_bf16(v16bf a, v16bf b, v8f c) {
    return __builtin_amdgcn_wmma_f32_16x16x32_bf16(
        /*neg_a=*/false, a, /*neg_b=*/false, b,
        /*c_mod=*/(short)0, c, /*reuse_a=*/false, /*reuse_b=*/false);
}

__device__ __forceinline__ uint32_t pack_bf16(float lo, float hi) {
    union { __bf16 b[2]; uint32_t u; } t;
    t.b[0] = (__bf16)lo; t.b[1] = (__bf16)hi;
    return t.u;
}

// Async 16B global->LDS copy (CDNA5 TDM-lite path, ASYNCcnt tracked).
// lds_addr: low 32 bits of the LDS destination address (per-lane);
// gbase: uniform 64-bit global base (SGPR pair); goff: per-lane byte offset.
__device__ __forceinline__ void async_copy_b128(uint32_t lds_addr,
                                                uint64_t gbase,
                                                uint32_t goff) {
    asm volatile("global_load_async_to_lds_b128 %0, %1, %2"
                 :: "v"(lds_addr), "v"(goff), "s"(gbase)
                 : "memory");
}
__device__ __forceinline__ void async_wait0() {
    asm volatile("s_wait_asynccnt 0x0" ::: "memory");
}

// A-matrix fragment (16x32 bf16) from row-major tile, leading dim ld.
// lanes 0-15: K = 0..15 ; lanes 16-31: K = 16..31. One contiguous 32B run.
__device__ __forceinline__ v16bf frag_a(const __bf16* base, int ld) {
    const int lane = threadIdx.x & 31;
    const int m    = lane & 15;
    const int half = lane >> 4;
    v16bf f;
#pragma unroll
    for (int v = 0; v < 8; ++v) {
        int k = half * 16 + 2 * v;
        f[2 * v]     = base[m * ld + k];
        f[2 * v + 1] = base[m * ld + k + 1];
    }
    return f;
}

// B-matrix fragment (32x16, K x N) from a TRANSPOSED tile stored [n][k].
// lanes 0-15 hold K=0..15, lanes 16-31 K=16..31: one contiguous 32B run.
__device__ __forceinline__ v16bf frag_bT(const __bf16* base, int ld) {
    const int lane = threadIdx.x & 31;
    const int n    = lane & 15;
    const int half = lane >> 4;
    v16bf f;
#pragma unroll
    for (int v = 0; v < 8; ++v) {
        int k = half * 16 + 2 * v;
        f[2 * v]     = base[n * ld + k];
        f[2 * v + 1] = base[n * ld + k + 1];
    }
    return f;
}

// ---------------------------------------------------------------------------
// Kernel 1: qkv = x @ Wqkv + bqkv -> bf16 Q/K/V in [B,H,S,D].
// Block tile 64(M) x 128(N), 8 waves; wave w owns columns [w*16, w*16+16).
// ---------------------------------------------------------------------------
__global__ __launch_bounds__(256) void qkv_gemm_kernel(
    const float* __restrict__ x, const float* __restrict__ Wqkv,
    const float* __restrict__ bqkv,
    __bf16* __restrict__ q_ws, __bf16* __restrict__ k_ws,
    __bf16* __restrict__ v_ws)
{
    __shared__ alignas(16) __bf16 ldsA[64 * LDA];    // [m][k]
    __shared__ alignas(16) __bf16 ldsBt[128 * LDA];  // [n][k] transposed

    const int tid  = threadIdx.x;
    const int wave = tid >> 5;
    const int lane = tid & 31;
    const int half = lane >> 4;
    const int n    = lane & 15;
    const int bm   = blockIdx.y * 64;
    const int bn   = blockIdx.x * 128;

    v8f acc[4] = {};

    for (int k0 = 0; k0 < D_MODEL; k0 += 32) {
        // A tile 64x32: f32 -> packed bf16 pairs (dword LDS stores)
        {
            uint32_t* la = (uint32_t*)ldsA;
            for (int i = tid; i < 64 * 16; i += 256) {
                int r = i >> 4, c2 = i & 15;
                const float* p = &x[(size_t)(bm + r) * D_MODEL + k0 + 2 * c2];
                la[r * (LDA / 2) + c2] = pack_bf16(p[0], p[1]);
            }
        }
        // B tile 32x128 stored transposed [n][k]
        for (int i = tid; i < 32 * 128; i += 256) {
            int r = i >> 7, c = i & 127;   // r = k, c = n (coalesced in n)
            ldsBt[c * LDA + r] =
                (__bf16)Wqkv[(size_t)(k0 + r) * (3 * D_MODEL) + bn + c];
        }
        // prefetch next K-step tiles
        if (k0 + 32 < D_MODEL) {
            const char* pw = (const char*)&Wqkv[(size_t)(k0 + 32) * (3 * D_MODEL) + bn];
            const char* pa = (const char*)&x[(size_t)bm * D_MODEL + k0 + 32];
            if (tid < 128)       __builtin_prefetch(pw + (size_t)tid * 128, 0, 0);
            else if (tid < 192)  __builtin_prefetch(pa + (size_t)(tid - 128) * 128, 0, 0);
        }
        __syncthreads();

        v16bf bf = frag_bT(ldsBt + wave * 16 * LDA, LDA);
#pragma unroll
        for (int mt = 0; mt < 4; ++mt) {
            v16bf af = frag_a(ldsA + mt * 16 * LDA, LDA);
            acc[mt] = wmma_bf16(af, bf, acc[mt]);
        }
        __syncthreads();
    }

    // Epilogue: bias + scatter into Q/K/V [B,H,S,D] as bf16.
    const int col  = bn + wave * 16 + n;          // 0..3071
    const int part = col / D_MODEL;               // 0=Q 1=K 2=V
    const int e    = col % D_MODEL;
    const int h    = e >> 6;
    const int d    = e & 63;
    __bf16* dst = (part == 0) ? q_ws : ((part == 1) ? k_ws : v_ws);
    const float bias = bqkv[col];

#pragma unroll
    for (int mt = 0; mt < 4; ++mt) {
#pragma unroll
        for (int v = 0; v < 8; ++v) {
            int m = bm + mt * 16 + v + 8 * half;  // global row = b*S + s
            int b = m >> 11;
            int s = m & (SEQ - 1);
            dst[(((size_t)(b * N_HEADS + h) * SEQ + s) * HEAD_DIM + d)] =
                (__bf16)(acc[mt][v] + bias);
        }
    }
}

// ---------------------------------------------------------------------------
// Kernel 2: causal flash attention. grid = (S/128, B*H). 8 waves, each wave
// owns a 16-row query tile resident as two A-fragments; K/V streamed in
// 32-row chunks (K via async global->LDS); online softmax via shfl.
// ---------------------------------------------------------------------------
__global__ __launch_bounds__(256) void attn_kernel(
    const __bf16* __restrict__ q_ws, const __bf16* __restrict__ k_ws,
    const __bf16* __restrict__ v_ws, __bf16* __restrict__ y_ws)
{
    __shared__ alignas(16) __bf16 ldsK[32 * LDK];      // [t][d] padded
    __shared__ alignas(16) __bf16 ldsVt[64 * LDA];     // [d][t] transposed
    __shared__ alignas(16) __bf16 ldsP[8][16 * LDA];   // wave-private P [m][t]

    const int tid  = threadIdx.x;
    const int wave = tid >> 5;
    const int lane = tid & 31;
    const int half = lane >> 4;
    const int n    = lane & 15;

    const int bh     = blockIdx.y;           // b*16 + h
    const int qblock = blockIdx.x * 128;
    const int q0     = qblock + wave * 16;

    const __bf16* qbase = q_ws + (size_t)bh * SEQ * HEAD_DIM;
    const __bf16* kbase = k_ws + (size_t)bh * SEQ * HEAD_DIM;
    const __bf16* vbase = v_ws + (size_t)bh * SEQ * HEAD_DIM;

    // Resident Q tile: 16 x 64 as two 16x32 A-fragments (global b128 loads).
    v16bf qf0 = frag_a(qbase + (size_t)q0 * HEAD_DIM, HEAD_DIM);
    v16bf qf1 = frag_a(qbase + (size_t)q0 * HEAD_DIM + 32, HEAD_DIM);

    v8f acc[4] = {};                         // 16 x 64 output accumulator
    float rm[8], rl[8];
#pragma unroll
    for (int v = 0; v < 8; ++v) { rm[v] = -1e30f; rl[v] = 0.f; }

    // per-lane LDS destination for the async K copy (32 rows x 8 x 16B)
    const uint32_t ldsK_base = (uint32_t)(uintptr_t)&ldsK[0];
    const uint32_t k_dst = ldsK_base
                         + (uint32_t)((tid >> 3) * (LDK * 2) + (tid & 7) * 16);
    const uint32_t k_off = (uint32_t)tid * 16;   // contiguous 4KB chunk

    const int nchunks = (qblock + 128) / 32; // causal upper bound for block
    for (int ch = 0; ch < nchunks; ++ch) {
        const int t0 = ch * 32;

        // K chunk: async global->LDS b128 (256 lanes x 16B = 4KB)
        async_copy_b128(k_dst,
                        (uint64_t)(uintptr_t)(kbase + (size_t)t0 * HEAD_DIM),
                        k_off);
        // V chunk: transposed register scatter [d][t]
        for (int i = tid; i < 32 * 64; i += 256) {
            int t = i >> 6, d = i & 63;      // coalesced in d
            ldsVt[d * LDA + t] = vbase[(size_t)(t0 + t) * HEAD_DIM + d];
        }
        // prefetch next K/V chunk (4KB each = 32 x 128B lines)
        if (ch + 1 < nchunks) {
            const char* pk = (const char*)(kbase + (size_t)(t0 + 32) * HEAD_DIM);
            const char* pv = (const char*)(vbase + (size_t)(t0 + 32) * HEAD_DIM);
            if (tid < 32)       __builtin_prefetch(pk + (size_t)tid * 128, 0, 0);
            else if (tid < 64)  __builtin_prefetch(pv + (size_t)(tid - 32) * 128, 0, 0);
        }
        async_wait0();
        __syncthreads();

        // Skip fully-masked chunks (wave-uniform -> EXEC stays all-ones).
        if (t0 <= q0 + 15) {
            // scores: S = Q K^T, two 16x16 tiles covering t0..t0+31
            v8f sc[2] = {};
#pragma unroll
            for (int t = 0; t < 2; ++t) {
                v16bf kf0 = frag_bT(ldsK + t * 16 * LDK, LDK);       // d 0..31
                sc[t] = wmma_bf16(qf0, kf0, sc[t]);
                v16bf kf1 = frag_bT(ldsK + t * 16 * LDK + 32, LDK);  // d 32..63
                sc[t] = wmma_bf16(qf1, kf1, sc[t]);
            }

            // scale; causal mask only on diagonal chunks
            const bool need_mask = (t0 + 31 > q0);
#pragma unroll
            for (int t = 0; t < 2; ++t) {
#pragma unroll
                for (int v = 0; v < 8; ++v) {
                    float s = sc[t][v] * 0.125f;   // 1/sqrt(64)
                    if (need_mask) {
                        int tg = t0 + t * 16 + n;
                        int qg = q0 + v + 8 * half;
                        s = (tg <= qg) ? s : -1e30f;
                    }
                    sc[t][v] = s;
                }
            }

            // online softmax
#pragma unroll
            for (int v = 0; v < 8; ++v) {
                float mx = fmaxf(sc[0][v], sc[1][v]);
                mx = fmaxf(mx, __shfl_xor(mx, 1, 32));
                mx = fmaxf(mx, __shfl_xor(mx, 2, 32));
                mx = fmaxf(mx, __shfl_xor(mx, 4, 32));
                mx = fmaxf(mx, __shfl_xor(mx, 8, 32));

                float mnew = fmaxf(rm[v], mx);
                float corr = __expf(rm[v] - mnew);
                rm[v] = mnew;

                float p0 = __expf(sc[0][v] - mnew);
                float p1 = __expf(sc[1][v] - mnew);
                sc[0][v] = p0; sc[1][v] = p1;

                float rs = p0 + p1;
                rs += __shfl_xor(rs, 1, 32);
                rs += __shfl_xor(rs, 2, 32);
                rs += __shfl_xor(rs, 4, 32);
                rs += __shfl_xor(rs, 8, 32);
                rl[v] = rl[v] * corr + rs;

#pragma unroll
                for (int dt = 0; dt < 4; ++dt) acc[dt][v] *= corr;
            }

            // C-layout -> A-layout for P via wave-private LDS bounce
            __bf16* P = &ldsP[wave][0];
#pragma unroll
            for (int t = 0; t < 2; ++t)
#pragma unroll
                for (int v = 0; v < 8; ++v)
                    P[(v + 8 * half) * LDA + t * 16 + n] = (__bf16)sc[t][v];

            v16bf pf = frag_a(P, LDA);

            // O += P @ V  (four 16x16 d-tiles; V^T tile rows are d)
#pragma unroll
            for (int dt = 0; dt < 4; ++dt) {
                v16bf vf = frag_bT(ldsVt + dt * 16 * LDA, LDA);
                acc[dt] = wmma_bf16(pf, vf, acc[dt]);
            }
        }
        __syncthreads();
    }

    // finalize, store y in [B, S, H, D] (row-major [B*S, 1024] for out GEMM)
    const int b = bh >> 4;
    const int h = bh & 15;
#pragma unroll
    for (int dt = 0; dt < 4; ++dt) {
#pragma unroll
        for (int v = 0; v < 8; ++v) {
            int s = q0 + v + 8 * half;
            int d = dt * 16 + n;
            float o = acc[dt][v] / rl[v];
            y_ws[(((size_t)(b * SEQ + s) * N_HEADS + h) * HEAD_DIM + d)] = (__bf16)o;
        }
    }
}

// ---------------------------------------------------------------------------
// Kernel 3: out = y @ Wo + bo (f32 output). A tile staged via async copy.
// ---------------------------------------------------------------------------
__global__ __launch_bounds__(256) void out_gemm_kernel(
    const __bf16* __restrict__ y_ws, const float* __restrict__ Wo,
    const float* __restrict__ bo, float* __restrict__ out)
{
    __shared__ alignas(16) __bf16 ldsA[64 * LDA];
    __shared__ alignas(16) __bf16 ldsBt[128 * LDA];

    const int tid  = threadIdx.x;
    const int wave = tid >> 5;
    const int lane = tid & 31;
    const int half = lane >> 4;
    const int n    = lane & 15;
    const int bm   = blockIdx.y * 64;
    const int bn   = blockIdx.x * 128;

    // per-lane async A-copy addressing: 64 rows x 4 x 16B = 4KB
    const uint32_t ldsA_base = (uint32_t)(uintptr_t)&ldsA[0];
    const uint32_t a_dst = ldsA_base
                         + (uint32_t)((tid >> 2) * (LDA * 2) + (tid & 3) * 16);
    const uint32_t a_off = (uint32_t)((tid >> 2) * (D_MODEL * 2) + (tid & 3) * 16);

    v8f acc[4] = {};

    for (int k0 = 0; k0 < D_MODEL; k0 += 32) {
        // A tile (bf16 already): async global->LDS b128, padded rows
        async_copy_b128(a_dst,
                        (uint64_t)(uintptr_t)(y_ws + (size_t)bm * D_MODEL + k0),
                        a_off);
        // B tile 32x128 stored transposed [n][k] (f32 -> bf16 on the fly)
        for (int i = tid; i < 32 * 128; i += 256) {
            int r = i >> 7, c = i & 127;
            ldsBt[c * LDA + r] = (__bf16)Wo[(size_t)(k0 + r) * D_MODEL + bn + c];
        }
        if (k0 + 32 < D_MODEL) {
            const char* pw = (const char*)&Wo[(size_t)(k0 + 32) * D_MODEL + bn];
            const char* pa = (const char*)(y_ws + (size_t)bm * D_MODEL + k0 + 32);
            if (tid < 128)       __builtin_prefetch(pw + (size_t)tid * 128, 0, 0);
            else if (tid < 160)  __builtin_prefetch(pa + (size_t)(tid - 128) * 128, 0, 0);
        }
        async_wait0();
        __syncthreads();

        v16bf bf = frag_bT(ldsBt + wave * 16 * LDA, LDA);
#pragma unroll
        for (int mt = 0; mt < 4; ++mt) {
            v16bf af = frag_a(ldsA + mt * 16 * LDA, LDA);
            acc[mt] = wmma_bf16(af, bf, acc[mt]);
        }
        __syncthreads();
    }

    const int   col  = bn + wave * 16 + n;
    const float bias = bo[col];
#pragma unroll
    for (int mt = 0; mt < 4; ++mt) {
#pragma unroll
        for (int v = 0; v < 8; ++v) {
            int m = bm + mt * 16 + v + 8 * half;
            out[(size_t)m * D_MODEL + col] = acc[mt][v] + bias;
        }
    }
}

// ---------------------------------------------------------------------------
extern "C" void kernel_launch(void* const* d_in, const int* in_sizes, int n_in,
                              void* d_out, int out_size, void* d_ws, size_t ws_size,
                              hipStream_t stream) {
    const float* x    = (const float*)d_in[0];
    const float* Wqkv = (const float*)d_in[1];
    const float* bqkv = (const float*)d_in[2];
    const float* Wo   = (const float*)d_in[3];
    const float* bo   = (const float*)d_in[4];
    float* out = (float*)d_out;

    const size_t elems = (size_t)BATCH * SEQ * D_MODEL;  // 8,388,608
    __bf16* q_ws = (__bf16*)d_ws;
    __bf16* k_ws = q_ws + elems;
    __bf16* v_ws = k_ws + elems;
    __bf16* y_ws = v_ws + elems;   // total 4*elems*2B = 64 MiB of d_ws

    // 1) QKV projection: M=8192, N=3072, K=1024; block tile 64x128
    qkv_gemm_kernel<<<dim3(3 * D_MODEL / 128, BATCH * SEQ / 64), 256, 0, stream>>>(
        x, Wqkv, bqkv, q_ws, k_ws, v_ws);

    // 2) Causal flash attention: grid (S/128, B*H)
    attn_kernel<<<dim3(SEQ / 128, BATCH * N_HEADS), 256, 0, stream>>>(
        q_ws, k_ws, v_ws, y_ws);

    // 3) Output projection: M=8192, N=1024, K=1024
    out_gemm_kernel<<<dim3(D_MODEL / 128, BATCH * SEQ / 64), 256, 0, stream>>>(
        y_ws, Wo, bo, out);
}
